// SelfAttention_33895881900719
// MI455X (gfx1250) — matrix operand order repaired and checked by reading, not compiled, not run on hardware
//
#include <hip/hip_runtime.h>
#include <hip/hip_bf16.h>
#include <stdint.h>

// ---------------------------------------------------------------------------
// Self-attention block for MI455X (gfx1250, wave32, WMMA + async LDS copies).
// B=8, C=256, N=4096, GROUPS=32.
// ---------------------------------------------------------------------------

#define BATCH   8
#define CDIM    256
#define NDIM    4096
#define NGROUP  32
#define QSCALE  0.0625f   /* 256^-0.5 */

typedef __attribute__((ext_vector_type(16))) __bf16 v16bf;
typedef __attribute__((ext_vector_type(8)))  float  v8f;

union Frag {
  v16bf v;
  uint4 u[2];
  unsigned short s[16];
};

__device__ __forceinline__ unsigned short f2bf(float f) {
  unsigned int u = __float_as_uint(f);
  u += 0x7FFFu + ((u >> 16) & 1u);     // round-to-nearest-even
  return (unsigned short)(u >> 16);
}

__device__ __forceinline__ v8f wmma_bf16(v16bf a, v16bf b, v8f c) {
  return __builtin_amdgcn_wmma_f32_16x16x32_bf16(false, a, false, b,
                                                 (short)0, c, false, false);
}

// Async global -> LDS copy, 16 bytes per lane, tracked by ASYNCcnt.
// VDST = per-lane LDS byte address (low 32 bits of generic pointer = AS3 offset).
__device__ __forceinline__ void async_ld16(const void* gaddr, const void* laddr) {
  const unsigned lds = (unsigned)(unsigned long long)laddr;
  asm volatile("global_load_async_to_lds_b128 %0, %1, off"
               :: "v"(lds), "v"(gaddr) : "memory");
}

// ---------------------------------------------------------------------------
// 1) GroupNorm statistics: one block per (b, group); 8 ch x 4096 px contiguous
// ---------------------------------------------------------------------------
__global__ void __launch_bounds__(256)
gn_stats_kernel(const float* __restrict__ x, float* __restrict__ stats) {
  const int bg = blockIdx.x;                       // b*32 + g
  const float* base = x + (size_t)bg * 32768;      // 8 channels * 4096
  float s = 0.f, ss = 0.f;
  for (int i = threadIdx.x; i < 32768; i += 256) {
    float v = base[i];
    s += v; ss += v * v;
  }
  for (int off = 1; off < 32; off <<= 1) {
    s  += __shfl_xor(s,  off, 32);
    ss += __shfl_xor(ss, off, 32);
  }
  __shared__ float rs[8], rss[8];
  const int wv = threadIdx.x >> 5, lane = threadIdx.x & 31;
  if (lane == 0) { rs[wv] = s; rss[wv] = ss; }
  __syncthreads();
  if (threadIdx.x == 0) {
    float S = 0.f, SS = 0.f;
    for (int i = 0; i < 8; ++i) { S += rs[i]; SS += rss[i]; }
    float mean = S * (1.0f / 32768.0f);
    float var  = SS * (1.0f / 32768.0f) - mean * mean;
    stats[bg * 2 + 0] = mean;
    stats[bg * 2 + 1] = rsqrtf(var + 1e-5f);
  }
}

// ---------------------------------------------------------------------------
// 2) fp32 -> bf16 weight conversion (qkv_weight 768x256, proj_weight 256x256)
// ---------------------------------------------------------------------------
__global__ void __launch_bounds__(256)
wconv_kernel(const float* __restrict__ wq, const float* __restrict__ wp,
             unsigned short* __restrict__ wqb, unsigned short* __restrict__ wpb) {
  const int i = blockIdx.x * 256 + threadIdx.x;
  if (i < 3 * CDIM * CDIM) wqb[i] = f2bf(wq[i]);
  const int j = i - 3 * CDIM * CDIM;
  if (j >= 0 && j < CDIM * CDIM) wpb[j] = f2bf(wp[j]);
}

// ---------------------------------------------------------------------------
// 3) Normalize + affine, write Hn in (B, N, C) bf16 (transposed for GEMM B-op)
// ---------------------------------------------------------------------------
__global__ void __launch_bounds__(256)
gn_apply_kernel(const float* __restrict__ x, const float* __restrict__ stats,
                const float* __restrict__ gw, const float* __restrict__ gb,
                unsigned short* __restrict__ Hn) {
  const int bc = blockIdx.x;            // b*256 + c
  const int b = bc >> 8, c = bc & 255;
  const int g = c >> 3;
  const float mean = stats[(b * NGROUP + g) * 2 + 0];
  const float rstd = stats[(b * NGROUP + g) * 2 + 1];
  const float w  = gw[c] * rstd;
  const float bi = gb[c] - mean * w;
  const float* xr = x + (size_t)bc * NDIM;
  unsigned short* hcol = Hn + (size_t)b * NDIM * CDIM + c;
  for (int n = threadIdx.x; n < NDIM; n += 256)
    hcol[(size_t)n * CDIM] = f2bf(xr[n] * w + bi);
}

// ---------------------------------------------------------------------------
// 4/6) GEMM: D[o,n] = sum_c W[o,c] * Act[n,c]  (K = 256 -> 8 wmma steps)
//   Each wave: 16(o) x 64(n) strip -> one A fragment feeds 4 wmma (32 total).
//   MODE 0 (QKV): split rows into Q (scaled, (B,N,C)), K ((B,N,C)), V ((B,C,N))
//   MODE 1 (PROJ): out = x + D + bias, fp32 (B,C,N)
// ---------------------------------------------------------------------------
template <int MODE>
__global__ void __launch_bounds__(128)
gemm_kernel(const unsigned short* __restrict__ Wbf,   // [M x 256] bf16
            const unsigned short* __restrict__ Act,   // [B, N, 256] bf16
            const float* __restrict__ bias,           // [M]
            unsigned short* __restrict__ Qt,
            unsigned short* __restrict__ Kt,
            unsigned short* __restrict__ Vc,
            const float* __restrict__ xres,
            float* __restrict__ outp) {
  const int lane = threadIdx.x & 31;
  const int wv   = threadIdx.x >> 5;
  const int col  = lane & 15;           // N-index within tile / A-row
  const int hi   = lane >> 4;           // half-wave select
  const int n_base = blockIdx.x * 256 + wv * 64;
  const int o_base = blockIdx.y * 16;
  const int b = blockIdx.z;

  v8f acc[4];
#pragma unroll
  for (int t = 0; t < 4; ++t) { v8f z = {}; acc[t] = z; }

  const unsigned short* wrow = Wbf + (size_t)(o_base + col) * CDIM;

#pragma unroll
  for (int kk = 0; kk < 8; ++kk) {
    Frag af;
    // A (16x32 bf16): lane<16 -> K = kk*32 + [0..15]; lane>=16 -> +16
    const unsigned short* ap = wrow + kk * 32 + hi * 16;
    af.u[0] = *(const uint4*)(ap);
    af.u[1] = *(const uint4*)(ap + 8);
#pragma unroll
    for (int t = 0; t < 4; ++t) {
      // B (32x16 bf16): lane fixed col, rows K = kk*32 + hi*16 + [0..15]
      Frag bf;
      const unsigned short* bp =
          Act + (size_t)(b * NDIM + n_base + t * 16 + col) * CDIM + kk * 32 + hi * 16;
      bf.u[0] = *(const uint4*)(bp);
      bf.u[1] = *(const uint4*)(bp + 8);
      acc[t] = wmma_bf16(af.v, bf.v, acc[t]);
    }
  }

#pragma unroll
  for (int t = 0; t < 4; ++t) {
#pragma unroll
    for (int r = 0; r < 8; ++r) {
      const int o = o_base + r + 8 * hi;     // D layout: M = r + 8*(lane/16)
      const int n = n_base + t * 16 + col;   //           N = lane%16
      float val = acc[t][r] + bias[o];
      if (MODE == 0) {
        if (o < CDIM) {                      // Q: pre-scale by C^-0.5
          Qt[(size_t)(b * NDIM + n) * CDIM + o] = f2bf(val * QSCALE);
        } else if (o < 2 * CDIM) {           // K
          Kt[(size_t)(b * NDIM + n) * CDIM + (o - CDIM)] = f2bf(val);
        } else {                             // V in (B, C, N)
          Vc[(size_t)(b * CDIM + (o - 2 * CDIM)) * NDIM + n] = f2bf(val);
        }
      } else {
        const size_t idx = (size_t)(b * CDIM + o) * NDIM + n;
        outp[idx] = xres[idx] + val;         // residual + proj_bias
      }
    }
  }
}

// ---------------------------------------------------------------------------
// 5) Flash attention: block = 4 waves, each wave owns one 16-row i-tile.
//    K/V j-chunks staged with double-buffered GLOBAL_LOAD_ASYNC_TO_LDS_B128
//    (ASYNCcnt), prefetching chunk n+1 while wmma pipeline runs on chunk n.
// ---------------------------------------------------------------------------
__global__ void __launch_bounds__(128)
attn_kernel(const unsigned short* __restrict__ Qt,   // (B,N,C) bf16, pre-scaled
            const unsigned short* __restrict__ Kt,   // (B,N,C) bf16
            const unsigned short* __restrict__ Vc,   // (B,C,N) bf16
            unsigned short* __restrict__ Ao) {       // (B,N,C) bf16
  __shared__ __align__(16) unsigned short qbuf[4][16 * CDIM];    // 32 KB
  __shared__ __align__(16) unsigned short kbuf[2][32 * CDIM];    // 32 KB (dbl)
  __shared__ __align__(16) unsigned short vbuf[2][CDIM * 32];    // 32 KB (dbl)
  __shared__ __align__(16) unsigned short pbuf[4][16 * 32];      //  4 KB

  const int tid  = threadIdx.x;
  const int lane = tid & 31;
  const int wv   = tid >> 5;
  const int col  = lane & 15;
  const int hi   = lane >> 4;
  const int b      = blockIdx.y;
  const int i_base = (blockIdx.x * 4 + wv) * 16;

  // Issue async stage of K rows / V columns for one 32-wide j-chunk.
  // 16 async instructions per wave per call.
  auto stage = [&](int sel, int j0) {
    const int row = tid >> 2, seg = tid & 3;          // K: 4 thr x 128B per row
    const unsigned short* gk = Kt + (size_t)(b * NDIM + j0 + row) * CDIM + seg * 64;
    unsigned short* lk = kbuf[sel] + row * CDIM + seg * 64;
#pragma unroll
    for (int q = 0; q < 8; ++q)
      async_ld16(gk + q * 8, lk + q * 8);
#pragma unroll
    for (int rr = 0; rr < 2; ++rr) {                  // V: 2 channel rows/thr
      const int c = tid * 2 + rr;
      const unsigned short* gv = Vc + (size_t)(b * CDIM + c) * NDIM + j0;
      unsigned short* lv = vbuf[sel] + c * 32;
#pragma unroll
      for (int q = 0; q < 4; ++q)
        async_ld16(gv + q * 8, lv + q * 8);
    }
  };

  // Stage this wave's Q tile (16 rows x 256) into LDS (one-time).
  {
    const int row = lane >> 1, seg = lane & 1;        // 2 lanes per 512B row
    const uint4* src = (const uint4*)(Qt + (size_t)(b * NDIM + i_base + row) * CDIM + seg * 128);
    uint4* dst = (uint4*)(qbuf[wv] + row * CDIM + seg * 128);
#pragma unroll
    for (int q = 0; q < 8; ++q) dst[q] = src[q];
  }

  v8f acc[16];
#pragma unroll
  for (int ct = 0; ct < 16; ++ct) { v8f z = {}; acc[ct] = z; }
  float mrow[8], lrow[8];
#pragma unroll
  for (int r = 0; r < 8; ++r) { mrow[r] = -1e30f; lrow[r] = 0.f; }

  stage(0, 0);                                        // prologue prefetch

  for (int j0 = 0; j0 < NDIM; j0 += 32) {
    const int cur = (j0 >> 5) & 1;
    if (j0 + 32 < NDIM) {
      stage(cur ^ 1, j0 + 32);                        // prefetch next chunk
      asm volatile("s_wait_asynccnt 16" ::: "memory"); // current chunk done
    } else {
      asm volatile("s_wait_asynccnt 0" ::: "memory");
    }
    __syncthreads();                                  // all waves' loads visible

    // --- S = Q K^T : two 16x16 tiles (j0..15, j16..31), K-dim = 256
    v8f s0 = {}, s1 = {};
#pragma unroll
    for (int kk = 0; kk < 8; ++kk) {
      Frag qf, b0, b1;
      const unsigned short* qp = qbuf[wv] + col * CDIM + kk * 32 + hi * 16;
      qf.u[0] = *(const uint4*)(qp);
      qf.u[1] = *(const uint4*)(qp + 8);
      const unsigned short* k0 = kbuf[cur] + (col)      * CDIM + kk * 32 + hi * 16;
      const unsigned short* k1 = kbuf[cur] + (16 + col) * CDIM + kk * 32 + hi * 16;
      b0.u[0] = *(const uint4*)(k0); b0.u[1] = *(const uint4*)(k0 + 8);
      b1.u[0] = *(const uint4*)(k1); b1.u[1] = *(const uint4*)(k1 + 8);
      s0 = wmma_bf16(qf.v, b0.v, s0);
      s1 = wmma_bf16(qf.v, b1.v, s1);
    }

    // --- online softmax (row i = r + 8*hi lives in a 16-lane group)
#pragma unroll
    for (int r = 0; r < 8; ++r) {
      float v0 = s0[r], v1 = s1[r];
      float mx = fmaxf(v0, v1);
      for (int off = 1; off < 16; off <<= 1) mx = fmaxf(mx, __shfl_xor(mx, off, 32));
      const float mnew  = fmaxf(mrow[r], mx);
      const float alpha = __expf(mrow[r] - mnew);
      const float p0 = __expf(v0 - mnew);
      const float p1 = __expf(v1 - mnew);
      float rs = p0 + p1;
      for (int off = 1; off < 16; off <<= 1) rs += __shfl_xor(rs, off, 32);
      lrow[r] = lrow[r] * alpha + rs;
      mrow[r] = mnew;
#pragma unroll
      for (int ct = 0; ct < 16; ++ct) acc[ct][r] *= alpha;
      const int irow = r + 8 * hi;                 // D-layout -> row-major P
      pbuf[wv][irow * 32 + col]      = f2bf(p0);
      pbuf[wv][irow * 32 + 16 + col] = f2bf(p1);
    }
    __syncthreads();   // make P visible across lanes (DS ordering)

    // --- acc += P (16x32) * V^T (32x256): 16 c-tiles
    Frag pf;
    const unsigned short* pp = pbuf[wv] + col * 32 + hi * 16;
    pf.u[0] = *(const uint4*)(pp);
    pf.u[1] = *(const uint4*)(pp + 8);
#pragma unroll
    for (int ct = 0; ct < 16; ++ct) {
      Frag vf;
      const unsigned short* vp = vbuf[cur] + (ct * 16 + col) * 32 + hi * 16;
      vf.u[0] = *(const uint4*)(vp);
      vf.u[1] = *(const uint4*)(vp + 8);
      acc[ct] = wmma_bf16(pf.v, vf.v, acc[ct]);
    }
    __syncthreads();   // reads of buf[cur] done before next prefetch overwrites
  }

  // --- normalize by row sum, store (B,N,C) bf16
#pragma unroll
  for (int r = 0; r < 8; ++r) {
    const float inv = 1.0f / lrow[r];
    const int i = i_base + r + 8 * hi;
    unsigned short* orow = Ao + (size_t)(b * NDIM + i) * CDIM;
#pragma unroll
    for (int ct = 0; ct < 16; ++ct)
      orow[ct * 16 + col] = f2bf(acc[ct][r] * inv);
  }
}

// ---------------------------------------------------------------------------
// Launch
// ---------------------------------------------------------------------------
extern "C" void kernel_launch(void* const* d_in, const int* in_sizes, int n_in,
                              void* d_out, int out_size, void* d_ws, size_t ws_size,
                              hipStream_t stream) {
  (void)in_sizes; (void)n_in; (void)out_size; (void)ws_size;
  const float* x     = (const float*)d_in[0];
  const float* gnw   = (const float*)d_in[1];
  const float* gnb   = (const float*)d_in[2];
  const float* qkvw  = (const float*)d_in[3];
  const float* qkvb  = (const float*)d_in[4];
  const float* projw = (const float*)d_in[5];
  const float* projb = (const float*)d_in[6];
  float* out = (float*)d_out;

  char* base = (char*)d_ws;
  size_t off = 0;
  auto carve = [&](size_t bytes) -> void* {
    void* p = base + off;
    off = (off + bytes + 255) & ~(size_t)255;
    return p;
  };
  float*          stats = (float*)carve((size_t)BATCH * NGROUP * 2 * sizeof(float));
  unsigned short* Wq    = (unsigned short*)carve((size_t)3 * CDIM * CDIM * 2);
  unsigned short* Wp    = (unsigned short*)carve((size_t)CDIM * CDIM * 2);
  unsigned short* Hn    = (unsigned short*)carve((size_t)BATCH * NDIM * CDIM * 2);
  unsigned short* Qt    = (unsigned short*)carve((size_t)BATCH * NDIM * CDIM * 2);
  unsigned short* Kt    = (unsigned short*)carve((size_t)BATCH * NDIM * CDIM * 2);
  unsigned short* Vc    = (unsigned short*)carve((size_t)BATCH * CDIM * NDIM * 2);
  unsigned short* Ao    = (unsigned short*)carve((size_t)BATCH * NDIM * CDIM * 2);

  gn_stats_kernel<<<BATCH * NGROUP, 256, 0, stream>>>(x, stats);
  wconv_kernel<<<(3 * CDIM * CDIM + CDIM * CDIM) / 256, 256, 0, stream>>>(qkvw, projw, Wq, Wp);
  gn_apply_kernel<<<BATCH * CDIM, 256, 0, stream>>>(x, stats, gnw, gnb, Hn);

  gemm_kernel<0><<<dim3(NDIM / 256, 3 * CDIM / 16, BATCH), 128, 0, stream>>>(
      Wq, Hn, qkvb, Qt, Kt, Vc, nullptr, nullptr);

  attn_kernel<<<dim3(NDIM / 64, BATCH), 128, 0, stream>>>(Qt, Kt, Vc, Ao);

  gemm_kernel<1><<<dim3(NDIM / 256, CDIM / 16, BATCH), 128, 0, stream>>>(
      Wp, Ao, projb, nullptr, nullptr, nullptr, x, out);
}
